// BinadamSSM_77103252898365
// MI455X (gfx1250) — compile-verified, matrix operand order checked by reading, not run on hardware
//
#include <hip/hip_runtime.h>

typedef __attribute__((ext_vector_type(16))) _Float16 v16h;
typedef __attribute__((ext_vector_type(8)))  _Float16 v8h;
typedef __attribute__((ext_vector_type(8)))  float    v8f;

#define BB 64
#define TT 512
#define II 128
#define HH 1024
#define OO 128

// ---------------------------------------------------------------------------
// Zero the state region of the workspace (h1 f32, h2 f32, S0, S1)
// ---------------------------------------------------------------------------
__global__ void zero_ws_kernel(unsigned int* __restrict__ p, int nwords) {
    int i = blockIdx.x * blockDim.x + threadIdx.x;
    if (i < nwords) p[i] = 0u;
}

// ---------------------------------------------------------------------------
// Build fused weight matrix Wcat[2048,1024] = [ s2 (.) Hf ; Wbn ] in f16,
// pre-swizzled into WMMA B-fragment order:
//   frag(nt,ks) holds the 32x16 B tile; per lane l, 16 contiguous halves.
//   lane l<16 : col n = nt*16+l,     halves q=0..7 -> k=ks*32+q,   q=8..15 -> k=ks*32+16+(q-8)
//   lane l>=16: col n = nt*16+l-16,  halves q=0..7 -> k=ks*32+8+q, q=8..15 -> k=ks*32+24+(q-8)
// Hf[i][j] = (-1)^popcount(i&j) / 32  (Hadamard(1024)/sqrt(1024)); s2 folded per column.
// ---------------------------------------------------------------------------
__global__ void __launch_bounds__(1024)
build_wcat_kernel(const float* __restrict__ s2, const float* __restrict__ Wbn,
                  _Float16* __restrict__ Wcat) {
    int id = blockIdx.x * blockDim.x + threadIdx.x;   // 0 .. 2^21-1
    int q  = id & 15;
    int l  = (id >> 4) & 31;
    int ks = (id >> 9) & 63;
    int nt = id >> 15;
    int n    = nt * 16 + (l & 15);
    int lsel = l >> 4;
    int kl   = (q < 8) ? (q + (lsel ? 8 : 0)) : ((q - 8) + (lsel ? 24 : 16));
    int k    = ks * 32 + kl;
    float v;
    if (k < HH) {
        float sgn = (__popc(k & n) & 1) ? -1.0f : 1.0f;
        v = sgn * (1.0f / 32.0f) * s2[n];
    } else {
        v = Wbn[(size_t)(k - HH) * HH + n];
    }
    Wcat[id] = (_Float16)v;
}

// ---------------------------------------------------------------------------
// Per-step layer 1: h1 = s1 * blockWHT32(h1) + b1 + x[:,t,:] @ W_in
// One wave per (batch row b, 32-block blk). FWHT via 5 shfl_xor butterflies.
// Writes relu(h1) in f16 into the h1b half of the S buffer read by step_h2(t).
// ---------------------------------------------------------------------------
__global__ void __launch_bounds__(1024)
step_h1_kernel(const float* __restrict__ x, const float* __restrict__ W_in,
               const float* __restrict__ s1, const float* __restrict__ b1,
               float* __restrict__ h1, _Float16* __restrict__ Swr, int t) {
    const int b   = blockIdx.x;          // 0..63
    const int tid = threadIdx.x;         // 0..1023
    const int j   = tid & 31;            // lane within 32-block
    const int h   = tid;                 // column index (blk*32 + j)

    __shared__ float xsh[II];
    if (tid < II) xsh[tid] = x[((size_t)b * TT + t) * II + tid];
    __syncthreads();

    // 32-point fast Walsh-Hadamard transform across the wave
    float v = h1[(size_t)b * HH + h];
#pragma unroll
    for (int s = 16; s >= 1; s >>= 1) {
        float p = __shfl_xor(v, s, 32);
        v = (j & s) ? (p - v) : (v + p);
    }
    v *= 0.17677669529663687f;           // 1/sqrt(32)

    // u_t[b,h] = x[b,t,:] . W_in[:,h]
    float u = 0.0f;
#pragma unroll 8
    for (int i = 0; i < II; ++i)
        u = fmaf(xsh[i], W_in[(size_t)i * HH + h], u);

    float hn = fmaf(s1[h], v, b1[h]) + u;
    h1[(size_t)b * HH + h] = hn;
    float r = hn > 0.0f ? hn : 0.0f;
    Swr[(size_t)b * 2048 + HH + h] = (_Float16)r;
}

// ---------------------------------------------------------------------------
// Per-step layer 2 (fused): h2_new[64,1024] = Srd[64,2048] @ Wcat[2048,1024] + b2
// where Srd = [ h2_f16 | relu(h1)_f16 ].
// 128 waves; each computes TWO N-adjacent 16x16 tiles (nt = 2*np, 2*np+1),
// sharing one A fragment per K-step and accumulating into two independent
// v8f accumulators (ILP across the WMMA pipe; halves A-fragment traffic).
// Writes f32 h2 (for the output head) and f16 h2 into the *other* S buffer.
// ---------------------------------------------------------------------------
__global__ void __launch_bounds__(128)
step_h2_kernel(const _Float16* __restrict__ Srd, _Float16* __restrict__ Swr,
               const _Float16* __restrict__ Wcat, const float* __restrict__ b2,
               float* __restrict__ h2) {
    const int lane = threadIdx.x & 31;
    const int wid  = blockIdx.x * 4 + (threadIdx.x >> 5);   // 0..127
    const int mt   = wid & 3;                                // M tile (0..3)
    const int np   = wid >> 2;                               // N tile pair (0..31)

    v8f c0 = {};
    v8f c1 = {};
    const int row = mt * 16 + (lane & 15);
    const _Float16* arow = Srd + (size_t)row * 2048 + ((lane >= 16) ? 8 : 0);
    const _Float16* bp0  = Wcat + ((size_t)(2 * np)     * 64 * 32 + lane) * 16;
    const _Float16* bp1  = Wcat + ((size_t)(2 * np + 1) * 64 * 32 + lane) * 16;

    for (int ks = 0; ks < 64; ++ks) {
        // A fragment: two contiguous 16B chunks per lane (documented 16-bit A layout)
        v8h alo = *(const v8h*)(arow + ks * 32);
        v8h ahi = *(const v8h*)(arow + ks * 32 + 16);
        v16h a = __builtin_shufflevector(alo, ahi, 0,1,2,3,4,5,6,7,8,9,10,11,12,13,14,15);

        // B fragments (pre-swizzled, 32 contiguous bytes per lane), two N tiles
        const _Float16* q0 = bp0 + (size_t)ks * 512;
        const _Float16* q1 = bp1 + (size_t)ks * 512;
        v8h b0lo = *(const v8h*)(q0);
        v8h b0hi = *(const v8h*)(q0 + 8);
        v16h bv0 = __builtin_shufflevector(b0lo, b0hi, 0,1,2,3,4,5,6,7,8,9,10,11,12,13,14,15);
        v8h b1lo = *(const v8h*)(q1);
        v8h b1hi = *(const v8h*)(q1 + 8);
        v16h bv1 = __builtin_shufflevector(b1lo, b1hi, 0,1,2,3,4,5,6,7,8,9,10,11,12,13,14,15);

        if (ks + 1 < 64) {
            __builtin_prefetch(q0 + 512, 0, 1);   // global_prefetch_b8
            __builtin_prefetch(q1 + 512, 0, 1);
        }

        // Two independent WMMAs sharing A -> no accumulator RAW between them
        c0 = __builtin_amdgcn_wmma_f32_16x16x32_f16(
                false, a, false, bv0, (short)0, c0, false, false);
        c1 = __builtin_amdgcn_wmma_f32_16x16x32_f16(
                false, a, false, bv1, (short)0, c1, false, false);
    }

    const int n0    = (2 * np) * 16 + (lane & 15);
    const int n1    = n0 + 16;
    const float bias0 = b2[n0];
    const float bias1 = b2[n1];
    const int mbase = mt * 16 + ((lane >= 16) ? 8 : 0);
#pragma unroll
    for (int r = 0; r < 8; ++r) {
        int m = mbase + r;
        float v0 = c0[r] + bias0;
        float v1 = c1[r] + bias1;
        h2[(size_t)m * HH + n0] = v0;                 // f32 state (output head)
        h2[(size_t)m * HH + n1] = v1;
        Swr[(size_t)m * 2048 + n0] = (_Float16)v0;    // f16 operand for next step
        Swr[(size_t)m * 2048 + n1] = (_Float16)v1;
    }
}

// ---------------------------------------------------------------------------
// Output head: out[b,o] = relu(h2[b,:]) . W_out[:,o] + b_out[o]
// ---------------------------------------------------------------------------
__global__ void __launch_bounds__(256)
out_proj_kernel(const float* __restrict__ h2, const float* __restrict__ W_out,
                const float* __restrict__ b_out, float* __restrict__ out) {
    int id = blockIdx.x * blockDim.x + threadIdx.x;   // 0..8191
    int b = id >> 7, o = id & 127;
    float acc = b_out[o];
    for (int hh = 0; hh < HH; ++hh) {
        float r = h2[(size_t)b * HH + hh];
        r = r > 0.0f ? r : 0.0f;
        acc = fmaf(r, W_out[(size_t)hh * OO + o], acc);
    }
    out[id] = acc;
}

// ---------------------------------------------------------------------------
// Host launcher
// ---------------------------------------------------------------------------
extern "C" void kernel_launch(void* const* d_in, const int* in_sizes, int n_in,
                              void* d_out, int out_size, void* d_ws, size_t ws_size,
                              hipStream_t stream) {
    const float* x     = (const float*)d_in[0];
    const float* W_in  = (const float*)d_in[1];
    const float* s1    = (const float*)d_in[2];
    const float* b1    = (const float*)d_in[3];
    const float* s2    = (const float*)d_in[4];
    const float* b2    = (const float*)d_in[5];
    const float* Wbn   = (const float*)d_in[6];
    const float* W_out = (const float*)d_in[7];
    const float* b_out = (const float*)d_in[8];
    float* out = (float*)d_out;

    char* ws = (char*)d_ws;
    float*     h1   = (float*)(ws);                     // 256 KB
    float*     h2   = (float*)(ws + 262144);            // 256 KB
    _Float16*  S0   = (_Float16*)(ws + 524288);         // 256 KB [64 x 2048]
    _Float16*  S1   = (_Float16*)(ws + 786432);         // 256 KB
    _Float16*  Wcat = (_Float16*)(ws + 1048576);        // 4 MB fused weights

    // Zero h1/h2/S0/S1 (first 1 MB = 262144 dwords)
    zero_ws_kernel<<<1024, 256, 0, stream>>>((unsigned int*)ws, 262144);

    // Build fused, pre-swizzled weight fragments (2^21 elements)
    build_wcat_kernel<<<2048, 1024, 0, stream>>>(s2, Wbn, Wcat);

    // Sequential scan: A(t) writes relu(h1) into S[t&1]; B(t) reads S[t&1],
    // writes h2_f16 into S[(t+1)&1].
    for (int t = 0; t < TT; ++t) {
        _Float16* Srd = (t & 1) ? S1 : S0;
        _Float16* Swr = (t & 1) ? S0 : S1;
        step_h1_kernel<<<BB, 1024, 0, stream>>>(x, W_in, s1, b1, h1, Srd, t);
        step_h2_kernel<<<32, 128, 0, stream>>>(Srd, Swr, Wcat, b2, h2);
    }

    out_proj_kernel<<<32, 256, 0, stream>>>(h2, W_out, b_out, out);
}
